// AdaptiveRoutingLayer_39041252721314
// MI455X (gfx1250) — matrix operand order repaired and checked by reading, not compiled, not run on hardware
//
#include <hip/hip_runtime.h>

#define NEXP 8
#define DDIM 2048
#define BTOK 8192

#define BM 128
#define BN 128
#define BK 32
#define LDT 40   // padded LDS row stride (elements) to avoid bank conflicts

typedef __attribute__((ext_vector_type(16))) __bf16 v16bf;
typedef __attribute__((ext_vector_type(8)))  float  v8f;
typedef __attribute__((ext_vector_type(4)))  int    v4i;

__device__ __forceinline__ unsigned short f32_to_bf16(float f) {
    unsigned int u = __float_as_uint(f);
    u += 0x7FFFu + ((u >> 16) & 1u);   // round-to-nearest-even
    return (unsigned short)(u >> 16);
}

__device__ __forceinline__ void softmax8(const float* __restrict__ rw, int t,
                                         float* __restrict__ w) {
    float r[NEXP];
    float mx = -3.4e38f;
#pragma unroll
    for (int e = 0; e < NEXP; ++e) { r[e] = rw[t * NEXP + e]; mx = fmaxf(mx, r[e]); }
    float s = 0.f;
#pragma unroll
    for (int e = 0; e < NEXP; ++e) { r[e] = __expf(r[e] - mx); s += r[e]; }
    float inv = 1.f / s;
#pragma unroll
    for (int e = 0; e < NEXP; ++e) w[e] = r[e] * inv;
}

// ---- CDNA5 async global->LDS copy (ASYNCcnt path) ------------------------
__device__ __forceinline__ void async_copy_b128(const unsigned short* g,
                                                unsigned short* l) {
#if defined(__gfx1250__) && __has_builtin(__builtin_amdgcn_global_load_async_to_lds_b128)
    __builtin_amdgcn_global_load_async_to_lds_b128(
        (__attribute__((address_space(1))) v4i*)(unsigned long long)g,
        (__attribute__((address_space(3))) v4i*)(unsigned int)(unsigned long long)l,
        0, 0);
#elif defined(__gfx1250__)
    unsigned int loff = (unsigned int)(unsigned long long)l;  // low 32b = LDS offset
    asm volatile("global_load_async_to_lds_b128 %0, %1, off"
                 :: "v"(loff), "v"((unsigned long long)g) : "memory");
#else
    ((uint4*)l)[0] = ((const uint4*)g)[0];   // host-pass parse fallback
#endif
}

__device__ __forceinline__ void wait_async0() {
#if defined(__gfx1250__) && __has_builtin(__builtin_amdgcn_s_wait_asynccnt)
    __builtin_amdgcn_s_wait_asynccnt(0);
#elif defined(__gfx1250__)
    asm volatile("s_wait_asynccnt 0x0" ::: "memory");
#endif
}

// ---- Kernel 1: combined bias b_comb[o] = sum_e w_e * b[e][o] -------------
__global__ void prep_bias_kernel(const float* __restrict__ b,
                                 const float* __restrict__ rw,
                                 const int* __restrict__ task,
                                 float* __restrict__ bcomb) {
    float w[NEXP];
    softmax8(rw, task[0], w);
    int o = blockIdx.x * blockDim.x + threadIdx.x;
    float acc = 0.f;
#pragma unroll
    for (int e = 0; e < NEXP; ++e) acc += w[e] * b[e * DDIM + o];
    bcomb[o] = acc;
}

// ---- Kernel 2: W_c(bf16) = sum_e w_e * W[e], streaming float4 ------------
__global__ void combine_w_kernel(const float* __restrict__ W,
                                 const float* __restrict__ rw,
                                 const int* __restrict__ task,
                                 unsigned short* __restrict__ wc) {
    float w[NEXP];
    softmax8(rw, task[0], w);
    size_t idx = ((size_t)blockIdx.x * blockDim.x + threadIdx.x) * 4;
    float ax = 0.f, ay = 0.f, az = 0.f, aw = 0.f;
#pragma unroll
    for (int e = 0; e < NEXP; ++e) {
        const float4 v = *(const float4*)(W + (size_t)e * DDIM * DDIM + idx);
        ax += w[e] * v.x; ay += w[e] * v.y; az += w[e] * v.z; aw += w[e] * v.w;
    }
    ushort4 o;
    o.x = f32_to_bf16(ax); o.y = f32_to_bf16(ay);
    o.z = f32_to_bf16(az); o.w = f32_to_bf16(aw);
    *(ushort4*)(wc + idx) = o;
}

// ---- Kernel 2b: x(bf16) = cvt(x), streaming ------------------------------
__global__ void cvt_x_kernel(const float* __restrict__ x,
                             unsigned short* __restrict__ xb) {
    size_t idx = ((size_t)blockIdx.x * blockDim.x + threadIdx.x) * 4;
    const float4 v = *(const float4*)(x + idx);
    ushort4 o;
    o.x = f32_to_bf16(v.x); o.y = f32_to_bf16(v.y);
    o.z = f32_to_bf16(v.z); o.w = f32_to_bf16(v.w);
    *(ushort4*)(xb + idx) = o;
}

// ---- shared WMMA tile compute -------------------------------------------
__device__ __forceinline__ void compute_tile(const unsigned short* tA,
                                             const unsigned short* tB,
                                             v8f (&acc)[4][2],
                                             int wm, int wn, int ln16, int hh) {
    union Frag { v16bf v; uint4 q[2]; };
    Frag bfrag[2];
#pragma unroll
    for (int fn = 0; fn < 2; ++fn) {
        const unsigned short* bp = tB + (wn * 32 + fn * 16 + ln16) * LDT + hh * 8;
        bfrag[fn].q[0] = *(const uint4*)(bp + 0);
        bfrag[fn].q[1] = *(const uint4*)(bp + 16);
    }
#pragma unroll
    for (int fm = 0; fm < 4; ++fm) {
        Frag afrag;
        const unsigned short* ap = tA + (wm * 64 + fm * 16 + ln16) * LDT;
        afrag.q[0] = *(const uint4*)(ap + hh * 8);
        afrag.q[1] = *(const uint4*)(ap + 16 + hh * 8);
#pragma unroll
        for (int fn = 0; fn < 2; ++fn) {
            acc[fm][fn] = __builtin_amdgcn_wmma_f32_16x16x32_bf16(
                false, afrag.v, false, bfrag[fn].v,
                (short)0, acc[fm][fn], false, false);
        }
    }
}

// ---- Kernel 3: out = x * W_c^T + b_comb, bf16 WMMA, double-buffered ------
template <bool ASYNC_A>
__global__ __launch_bounds__(256)
void moe_gemm_kernel(const unsigned short* __restrict__ xa,  // bf16 x (ASYNC_A)
                     const float* __restrict__ xf,           // f32  x (!ASYNC_A)
                     const unsigned short* __restrict__ wc,
                     const float* __restrict__ bcomb,
                     float* __restrict__ out) {
    __shared__ unsigned short ldsA[2][BM * LDT];
    __shared__ unsigned short ldsB[2][BN * LDT];

    const int tid  = threadIdx.x;
    const int lane = tid & 31;
    const int wave = tid >> 5;
    const int wm   = wave & 1;
    const int wn   = wave >> 1;
    const int ln16 = lane & 15;
    const int hh   = lane >> 4;

    const int row0 = blockIdx.y * BM;
    const int col0 = blockIdx.x * BN;

    const int lrow = tid >> 1;         // 0..127
    const int lk   = (tid & 1) * 16;   // 0 or 16

    v8f acc[4][2];
#pragma unroll
    for (int i = 0; i < 4; ++i)
#pragma unroll
        for (int j = 0; j < 2; ++j) acc[i][j] = (v8f){};

    float4 aregs[4];

    auto loadB = [&](int buf, int kk) {
        const unsigned short* src = wc + (size_t)(col0 + lrow) * DDIM + kk + lk;
        unsigned short* dst = &ldsB[buf][lrow * LDT + lk];
        async_copy_b128(src + 0, dst + 0);
        async_copy_b128(src + 8, dst + 8);
    };
    auto loadA_issue = [&](int buf, int kk) {
        if constexpr (ASYNC_A) {
            const unsigned short* src = xa + (size_t)(row0 + lrow) * DDIM + kk + lk;
            unsigned short* dst = &ldsA[buf][lrow * LDT + lk];
            async_copy_b128(src + 0, dst + 0);
            async_copy_b128(src + 8, dst + 8);
        } else {
            const float* src = xf + (size_t)(row0 + lrow) * DDIM + kk + lk;
#pragma unroll
            for (int j = 0; j < 4; ++j) aregs[j] = *(const float4*)(src + 4 * j);
        }
    };
    auto loadA_commit = [&](int buf) {
        if constexpr (!ASYNC_A) {
            alignas(16) unsigned short tmp[16];
#pragma unroll
            for (int j = 0; j < 4; ++j) {
                tmp[4 * j + 0] = f32_to_bf16(aregs[j].x);
                tmp[4 * j + 1] = f32_to_bf16(aregs[j].y);
                tmp[4 * j + 2] = f32_to_bf16(aregs[j].z);
                tmp[4 * j + 3] = f32_to_bf16(aregs[j].w);
            }
            *(uint4*)(&ldsA[buf][lrow * LDT + lk + 0]) = *(const uint4*)(&tmp[0]);
            *(uint4*)(&ldsA[buf][lrow * LDT + lk + 8]) = *(const uint4*)(&tmp[8]);
        }
    };

    // prologue: fill buffer 0
    loadA_issue(0, 0);
    loadB(0, 0);
    loadA_commit(0);
    wait_async0();
    __syncthreads();

    int buf = 0;
    for (int kk = 0; kk < DDIM; kk += BK) {
        const int nxt = buf ^ 1;
        const bool has_next = (kk + BK) < DDIM;
        if (has_next) {             // issue copies for next tile (ASYNCcnt)
            loadA_issue(nxt, kk + BK);
            loadB(nxt, kk + BK);
        }
        compute_tile(ldsA[buf], ldsB[buf], acc, wm, wn, ln16, hh);
        if (has_next) {
            loadA_commit(nxt);
            wait_async0();          // my async copies landed in LDS
        }
        __syncthreads();            // everyone done reading buf / writing nxt
        buf = nxt;
    }

    // epilogue: D[r][lane] holds M = r + 8*half, N = lane&15
#pragma unroll
    for (int fm = 0; fm < 4; ++fm) {
#pragma unroll
        for (int fn = 0; fn < 2; ++fn) {
            const int cg = col0 + wn * 32 + fn * 16 + ln16;
            const float bb = bcomb[cg];
#pragma unroll
            for (int r = 0; r < 8; ++r) {
                const int rg = row0 + wm * 64 + fm * 16 + hh * 8 + r;
                out[(size_t)rg * DDIM + cg] = acc[fm][fn][r] + bb;
            }
        }
    }
}

extern "C" void kernel_launch(void* const* d_in, const int* in_sizes, int n_in,
                              void* d_out, int out_size, void* d_ws, size_t ws_size,
                              hipStream_t stream) {
    const float* x    = (const float*)d_in[0];
    const float* W    = (const float*)d_in[1];
    const float* b    = (const float*)d_in[2];
    const float* rw   = (const float*)d_in[3];
    const int*   task = (const int*)d_in[4];
    float* out = (float*)d_out;

    float* bcomb = (float*)d_ws;                                  // 2048 f32
    unsigned short* wcomb =
        (unsigned short*)((char*)d_ws + DDIM * sizeof(float));    // 8 MB bf16
    const size_t off_xb = DDIM * sizeof(float) + (size_t)DDIM * DDIM * 2;
    unsigned short* xb = (unsigned short*)((char*)d_ws + off_xb); // 32 MB bf16
    const size_t need = off_xb + (size_t)BTOK * DDIM * 2;

    prep_bias_kernel<<<DDIM / 256, 256, 0, stream>>>(b, rw, task, bcomb);

    const int combine_blocks = (DDIM * DDIM) / (256 * 4);         // 4096
    combine_w_kernel<<<combine_blocks, 256, 0, stream>>>(W, rw, task, wcomb);

    dim3 grid(DDIM / BN, BTOK / BM);                              // 16 x 64
    if (ws_size >= need) {
        const int cvt_blocks = (int)(((size_t)BTOK * DDIM) / (256 * 4)); // 16384
        cvt_x_kernel<<<cvt_blocks, 256, 0, stream>>>(x, xb);
        moe_gemm_kernel<true><<<grid, 256, 0, stream>>>(xb, x, wcomb, bcomb, out);
    } else {
        moe_gemm_kernel<false><<<grid, 256, 0, stream>>>(nullptr, x, wcomb, bcomb, out);
    }
}
